// MultiHeadAttentionV2_81741817578291
// MI455X (gfx1250) — matrix operand order repaired and checked by reading, not compiled, not run on hardware
//
#include <hip/hip_runtime.h>
#include <hip/hip_bf16.h>

#define B_   2
#define SQ_  2048
#define SK_  2048
#define D_   1024
#define H_   16
#define HD_  64
#define NEGV (-1e9f)

// LDS score strip for fused attention
#define S_STRIDE    (SK_ + 8)                      // 2056 floats
#define SCRATCH_F   (4 * 8 * 32)                   // 1024 floats
#define SMEM_FLOATS (16 * S_STRIDE + SCRATCH_F)    // 33920
#define SMEM_BYTES  (SMEM_FLOATS * 4)              // 135680 B (< 320KB LDS/WGP)

typedef __attribute__((ext_vector_type(16))) __bf16 v16bf;
typedef __attribute__((ext_vector_type(8)))  float  v8f;
typedef __attribute__((ext_vector_type(4)))  __bf16 v4bf;
typedef __attribute__((ext_vector_type(4)))  unsigned int v4u;
typedef __attribute__((ext_vector_type(8)))  int v8i;
typedef __attribute__((ext_vector_type(4)))  int v4i;

#if defined(__gfx1250__) && __has_builtin(__builtin_amdgcn_tensor_load_to_lds) && \
    __has_builtin(__builtin_amdgcn_s_wait_tensorcnt)
#define USE_TDM 1
#else
#define USE_TDM 0
#endif

// ---------------------------------------------------------------------------
// WMMA helpers: wave32 16x16x32 bf16 -> f32 fragment layouts (ISA 7.12.2)
// ---------------------------------------------------------------------------
__device__ __forceinline__ v8f wmma_bf16(v16bf a, v16bf b, v8f c) {
    return __builtin_amdgcn_wmma_f32_16x16x32_bf16(
        false, a, false, b, (short)0, c, false, false);
}

__device__ __forceinline__ v16bf load_a_bf16(const __bf16* __restrict__ A, size_t lda,
                                             int m0, int k0, int lane) {
    const int m  = m0 + (lane & 15);
    const int kh = (lane >> 4) << 3;          // 0 or 8
    const __bf16* p = A + (size_t)m * lda + (size_t)(k0 + kh);
    v16bf a;
#pragma unroll
    for (int e = 0; e < 8; ++e) { a[e] = p[e]; a[e + 8] = p[e + 16]; }
    return a;
}

// A-fragment from a 16 x D_ bf16 strip resident in LDS
__device__ __forceinline__ v16bf load_a_ldsbf(const __bf16* S, int k0, int lane) {
    const int m  = lane & 15;
    const int kh = (lane >> 4) << 3;
    const __bf16* p = S + (size_t)m * D_ + k0 + kh;
    v16bf a;
#pragma unroll
    for (int e = 0; e < 8; ++e) { a[e] = p[e]; a[e + 8] = p[e + 16]; }
    return a;
}

// A-fragment from fp32 scores resident in LDS (cvt to bf16 on read)
__device__ __forceinline__ v16bf load_a_lds(const float* S, int k0, int lane) {
    const int m  = lane & 15;
    const int kh = (lane >> 4) << 3;
    const float* p = S + (size_t)m * S_STRIDE + k0 + kh;
    v16bf a;
#pragma unroll
    for (int e = 0; e < 8; ++e) { a[e] = (__bf16)p[e]; a[e + 8] = (__bf16)p[e + 16]; }
    return a;
}

// B[k][n] = Bt[n*ldt + k]  (operand stored N-major): two 16-byte runs/lane.
__device__ __forceinline__ v16bf load_b_transposed(const __bf16* __restrict__ Bt, size_t ldt,
                                                   int k0, int n0, int lane) {
    const int n  = n0 + (lane & 15);
    const int kk = k0 + ((lane >> 4) << 3);   // 0 or 8
    const __bf16* p = Bt + (size_t)n * ldt + kk;
    v16bf b;
#pragma unroll
    for (int e = 0; e < 8; ++e) { b[e] = p[e]; b[e + 8] = p[e + 16]; }
    return b;
}

// ---------------------------------------------------------------------------
// TDM: 1D contiguous strip, global -> LDS (ISA ch.8 D# layout).
// Low 32 bits of a generic LDS pointer are the LDS byte address (ISA 10.2).
// ---------------------------------------------------------------------------
#if USE_TDM
__device__ __forceinline__ void tdm_load_strip(const void* gsrc, unsigned lds_addr,
                                               unsigned nbytes) {
    const unsigned long long ga = (unsigned long long)gsrc;
    const unsigned elems = nbytes >> 1;           // data_size = 2 bytes
    v4u g0;
    g0[0] = 1u;                                   // count=1, user descriptor
    g0[1] = lds_addr;                             // lds_addr (bytes)
    g0[2] = (unsigned)(ga & 0xFFFFFFFFu);         // global_addr[31:0]
    g0[3] = (unsigned)((ga >> 32) & 0x01FFFFFFu)  // global_addr[56:32]
          | (2u << 30);                           // type=2 ("image")
    v8i g1;
    g1[0] = (int)(1u << 16);                      // data_size=1 -> 2B elements
    g1[1] = (int)((elems & 0xFFFFu) << 16);       // tensor_dim0[15:0]
    g1[2] = (int)(((elems >> 16) & 0xFFFFu)       // tensor_dim0[31:16]
          | (1u << 16));                          // tensor_dim1 = 1
    g1[3] = (int)((elems & 0xFFFFu) << 16);       // tile_dim0 = elems (1D)
    g1[4] = 0;                                    // tile_dim1/2 unused
    g1[5] = (int)elems;                           // tensor_dim0_stride
    g1[6] = 0;
    g1[7] = 0;
    v4i g2 = {0, 0, 0, 0}, g3 = {0, 0, 0, 0};     // <=2D: groups 2/3 unused
    v8i g4 = {0, 0, 0, 0, 0, 0, 0, 0};            // 6-arg toolchain extra group
    __builtin_amdgcn_tensor_load_to_lds(g0, g1, g2, g3, g4, 0);
}
#endif

// Stage a contiguous 16 x D_ bf16 A-strip (32 KB) into LDS, block-wide.
__device__ __forceinline__ void stage_a_strip(const __bf16* src, __bf16* Atile) {
#if USE_TDM
    if (threadIdx.y == 0) {                       // one TDM op per block
        tdm_load_strip((const void*)src,
                       (unsigned)(unsigned long long)(const void*)Atile,
                       16 * D_ * 2);
        __builtin_amdgcn_s_wait_tensorcnt(0);
    }
#else
    const int tid = threadIdx.y * 32 + threadIdx.x;
    const int4* s = (const int4*)src;
    int4* d = (int4*)Atile;
    for (int i = tid; i < (16 * D_ * 2) / 16; i += 128) d[i] = s[i];
#endif
    __syncthreads();
}

// ---------------------------------------------------------------------------
// Stage 0a: f32 -> bf16 elementwise conversion (vectorized x4)
// ---------------------------------------------------------------------------
__global__ void __launch_bounds__(256)
cvt_f32_bf16_kernel(const float* __restrict__ in, __bf16* __restrict__ out, int n4) {
    int i = blockIdx.x * 256 + threadIdx.x;
    if (i < n4) {
        const float4 v = ((const float4*)in)[i];
        v4bf o;
        o[0] = (__bf16)v.x; o[1] = (__bf16)v.y; o[2] = (__bf16)v.z; o[3] = (__bf16)v.w;
        ((v4bf*)out)[i] = o;
    }
}

// ---------------------------------------------------------------------------
// Stage 0b: f32 DxD -> bf16 transposed DxD (LDS-tiled, coalesced both sides)
// ---------------------------------------------------------------------------
__global__ void __launch_bounds__(1024)
transpose_cvt_kernel(const float* __restrict__ in, __bf16* __restrict__ outT) {
    __shared__ float t[32][33];
    const int bx = blockIdx.x * 32, by = blockIdx.y * 32;
    const int tx = threadIdx.x, ty = threadIdx.y;
    t[ty][tx] = in[(size_t)(by + ty) * D_ + (bx + tx)];
    __syncthreads();
    outT[(size_t)(bx + ty) * D_ + (by + tx)] = (__bf16)t[tx][ty];
}

// ---------------------------------------------------------------------------
// Stage 1: QKV projection.  out = (X @ W + bias) * scale, W pre-transposed.
// Block of 4 waves shares one 16-row A strip, TDM-staged into LDS.
// Each wave: 16x64 strip; 1 LDS A-frag + 4 global B-frags + 4 WMMA / k-step.
// ---------------------------------------------------------------------------
__global__ void __launch_bounds__(128)
proj_qkv_kernel(const __bf16* __restrict__ X, const __bf16* __restrict__ Wt,
                const float* __restrict__ bias, __bf16* __restrict__ outHead,
                float scale, int transposeOut) {
    __shared__ __bf16 Atile[16 * D_];               // 32 KB
    const int lane = threadIdx.x;
    const int nGroups = D_ / 64;                    // 16
    const int tile = blockIdx.x * 4 + threadIdx.y;
    const int m0  = (tile / nGroups) * 16;          // same for all 4 waves
    const int ng0 = (tile % nGroups) * 64;

    stage_a_strip(X + (size_t)m0 * D_, Atile);

    v8f acc[4] = {{}, {}, {}, {}};
    for (int k0 = 0; k0 < D_; k0 += 32) {
        __builtin_prefetch(Wt + (size_t)(ng0 + (lane & 15)) * D_ + k0 + 64, 0, 0);
        v16bf a = load_a_ldsbf(Atile, k0, lane);
#pragma unroll
        for (int j = 0; j < 4; ++j) {
            v16bf b = load_b_transposed(Wt, D_, k0, ng0 + 16 * j, lane);
            acc[j] = wmma_bf16(a, b, acc[j]);
        }
    }

#pragma unroll
    for (int j = 0; j < 4; ++j) {
        const int n  = ng0 + 16 * j + (lane & 15);
        const int h  = n >> 6;
        const int hd = n & 63;
        const float bv = bias[n];
#pragma unroll
        for (int r = 0; r < 8; ++r) {
            const int m  = m0 + r + ((lane >> 4) << 3);
            const int bb = m / SQ_;
            const int s  = m % SQ_;
            const float v = (acc[j][r] + bv) * scale;
            if (transposeOut) {   // (B,H,HD,S) for V
                outHead[((((size_t)bb * H_) + h) * HD_ + hd) * SQ_ + s] = (__bf16)v;
            } else {              // (B,H,S,HD) for Q,K
                outHead[((((size_t)bb * H_) + h) * SQ_ + s) * HD_ + hd] = (__bf16)v;
            }
        }
    }
}

// ---------------------------------------------------------------------------
// Stages 2-4 fused: scores -> LDS, softmax in LDS, weights NT-stored once,
// AV from LDS.  One block (8 waves, 256 thr) per (bh, 16-query strip).
// ---------------------------------------------------------------------------
__global__ void __launch_bounds__(256)
attn_fused_kernel(const __bf16* __restrict__ Q, const __bf16* __restrict__ K,
                  const __bf16* __restrict__ Vt, const float* __restrict__ mask,
                  float* __restrict__ weights, __bf16* __restrict__ attnOut) {
    extern __shared__ float smem[];
    float* Sld     = smem;                       // [16][S_STRIDE]
    float* scratch = smem + 16 * S_STRIDE;       // [4][8][32]

    const int lane = threadIdx.x & 31;
    const int wave = threadIdx.x >> 5;           // 0..7
    const int bh   = blockIdx.y;
    const int q0   = blockIdx.x * 16;
    const int b    = bh / H_;
    const int h    = bh % H_;

    const __bf16* Qb = Q  + (size_t)bh * SQ_ * HD_;
    const __bf16* Kb = K  + (size_t)bh * SK_ * HD_;
    const __bf16* Vb = Vt + (size_t)bh * HD_ * SK_;   // (HD, SK)
    const float* Mrow = mask + (size_t)b * SQ_ * SK_; // (B,1,SQ,SK)
    float* Wout = weights + (size_t)bh * SQ_ * SK_;

    // ---- Phase 1: scores = Q@K^T + mask -> LDS -------------------------
    const v16bf a0 = load_a_bf16(Qb, HD_, q0, 0,  lane);
    const v16bf a1 = load_a_bf16(Qb, HD_, q0, 32, lane);

#pragma unroll
    for (int pass = 0; pass < 4; ++pass) {
        const int n0 = (pass * 8 + wave) * 64;   // this wave's 64-key group
#pragma unroll
        for (int j = 0; j < 4; ++j) {
            v8f acc = {};
            acc = wmma_bf16(a0, load_b_transposed(Kb, HD_, 0,  n0 + 16 * j, lane), acc);
            acc = wmma_bf16(a1, load_b_transposed(Kb, HD_, 32, n0 + 16 * j, lane), acc);
            const int kcol = n0 + 16 * j + (lane & 15);
#pragma unroll
            for (int r = 0; r < 8; ++r) {
                const int rr = r + ((lane >> 4) << 3);
                const float mv = Mrow[(size_t)(q0 + rr) * SK_ + kcol];
                Sld[rr * S_STRIDE + kcol] = acc[r] + (1.0f - mv) * NEGV;
            }
        }
    }
    __syncthreads();

    // ---- Phase 2: softmax per row in LDS; NT-store normalized weights --
    {
        const int row = threadIdx.x >> 4;        // 0..15
        const int sub = threadIdx.x & 15;        // 16 threads per row
        float* Srow = Sld + row * S_STRIDE;
        float* Grow = Wout + (size_t)(q0 + row) * SK_;

        float lmax = -3.4e38f;
#pragma unroll 4
        for (int i = 0; i < SK_ / 16; ++i)
            lmax = fmaxf(lmax, Srow[sub + 16 * i]);
#pragma unroll
        for (int m = 8; m > 0; m >>= 1)
            lmax = fmaxf(lmax, __shfl_xor(lmax, m, 16));

        float lsum = 0.0f;
#pragma unroll 4
        for (int i = 0; i < SK_ / 16; ++i) {
            const float v = __expf(Srow[sub + 16 * i] - lmax);
            Srow[sub + 16 * i] = v;
            lsum += v;
        }
#pragma unroll
        for (int m = 8; m > 0; m >>= 1)
            lsum += __shfl_xor(lsum, m, 16);

        const float inv = 1.0f / lsum;
#pragma unroll 4
        for (int i = 0; i < SK_ / 16; ++i) {
            const float v = Srow[sub + 16 * i] * inv;
            Srow[sub + 16 * i] = v;
            __builtin_nontemporal_store(v, &Grow[sub + 16 * i]);  // write-once
        }
    }
    __syncthreads();

    // ---- Phase 3: attn strip = softmax(scores) @ V, A from LDS ---------
    {
        const int j     = wave & 3;              // HD tile 0..3
        const int khalf = wave >> 2;             // 0 or 1
        v8f acc = {};
        const int kbeg = khalf * (SK_ / 2);
        for (int k0 = kbeg; k0 < kbeg + SK_ / 2; k0 += 32) {
            v16bf a = load_a_lds(Sld, k0, lane);
            v16bf bfr = load_b_transposed(Vb, SK_, k0, 16 * j, lane);
            acc = wmma_bf16(a, bfr, acc);
        }
        if (khalf == 1) {
#pragma unroll
            for (int r = 0; r < 8; ++r)
                scratch[(j * 8 + r) * 32 + lane] = acc[r];
        }
        __syncthreads();
        if (khalf == 0) {
            const int hd = 16 * j + (lane & 15);
#pragma unroll
            for (int r = 0; r < 8; ++r) {
                const float v = acc[r] + scratch[(j * 8 + r) * 32 + lane];
                const int q = q0 + r + ((lane >> 4) << 3);
                attnOut[((size_t)b * SQ_ + q) * D_ + h * HD_ + hd] = (__bf16)v;
            }
        }
    }
}

// ---------------------------------------------------------------------------
// Stage 5: out = attn @ Wo + bo (fp32 out).  Wo pre-transposed.
// Block shares one TDM-staged A strip in LDS (same scheme as stage 1).
// ---------------------------------------------------------------------------
__global__ void __launch_bounds__(128)
oproj_kernel(const __bf16* __restrict__ Ain, const __bf16* __restrict__ WoT,
             const float* __restrict__ bias, float* __restrict__ out) {
    __shared__ __bf16 Atile[16 * D_];               // 32 KB
    const int lane = threadIdx.x;
    const int nGroups = D_ / 64;
    const int tile = blockIdx.x * 4 + threadIdx.y;
    const int m0  = (tile / nGroups) * 16;
    const int ng0 = (tile % nGroups) * 64;

    stage_a_strip(Ain + (size_t)m0 * D_, Atile);

    v8f acc[4] = {{}, {}, {}, {}};
    for (int k0 = 0; k0 < D_; k0 += 32) {
        __builtin_prefetch(WoT + (size_t)(ng0 + (lane & 15)) * D_ + k0 + 64, 0, 0);
        v16bf a = load_a_ldsbf(Atile, k0, lane);
#pragma unroll
        for (int j = 0; j < 4; ++j) {
            v16bf b = load_b_transposed(WoT, D_, k0, ng0 + 16 * j, lane);
            acc[j] = wmma_bf16(a, b, acc[j]);
        }
    }

#pragma unroll
    for (int j = 0; j < 4; ++j) {
        const int n = ng0 + 16 * j + (lane & 15);
        const float bv = bias[n];
#pragma unroll
        for (int r = 0; r < 8; ++r) {
            const int m = m0 + r + ((lane >> 4) << 3);
            out[(size_t)m * D_ + n] = acc[j][r] + bv;
        }
    }
}

// ---------------------------------------------------------------------------
extern "C" void kernel_launch(void* const* d_in, const int* in_sizes, int n_in,
                              void* d_out, int out_size, void* d_ws, size_t ws_size,
                              hipStream_t stream) {
    const float* hs   = (const float*)d_in[0];
    const float* kst  = (const float*)d_in[1];
    const float* vst  = (const float*)d_in[2];
    const float* mask = (const float*)d_in[3];
    const float* Wq = (const float*)d_in[4];  const float* bq = (const float*)d_in[5];
    const float* Wk = (const float*)d_in[6];  const float* bk = (const float*)d_in[7];
    const float* Wv = (const float*)d_in[8];  const float* bv = (const float*)d_in[9];
    const float* Wo = (const float*)d_in[10]; const float* bo = (const float*)d_in[11];

    float* out     = (float*)d_out;                         // (B,SQ,D)
    float* weights = out + (size_t)B_ * SQ_ * D_;           // (B,H,SQ,SK)

    // Workspace carve-up (bf16 elements)
    const size_t nW  = (size_t)D_ * D_;                     // 1M
    const size_t nX  = (size_t)B_ * SQ_ * D_;               // 4M
    const size_t nHQ = (size_t)B_ * H_ * SQ_ * HD_;         // 4M
    __bf16* p = (__bf16*)d_ws;
    __bf16* WqT = p;  p += nW;
    __bf16* WkT = p;  p += nW;
    __bf16* WvT = p;  p += nW;
    __bf16* WoT = p;  p += nW;
    __bf16* X_bf  = p;  p += nX;
    __bf16* KS_bf = p;  p += nX;
    __bf16* VS_bf = p;  p += nX;
    __bf16* Q_bf  = p;  p += nHQ;   // (B,H,SQ,HD)
    __bf16* K_bf  = p;  p += nHQ;   // (B,H,SK,HD)
    __bf16* Vt_bf = p;  p += nHQ;   // (B,H,HD,SK) transposed
    __bf16* A_bf  = p;  p += nX;    // (B,SQ,D)

    // Stage 0: conversions (+ weight transposes)
    {
        dim3 tg(D_ / 32, D_ / 32), tb(32, 32);
        transpose_cvt_kernel<<<tg, tb, 0, stream>>>(Wq, WqT);
        transpose_cvt_kernel<<<tg, tb, 0, stream>>>(Wk, WkT);
        transpose_cvt_kernel<<<tg, tb, 0, stream>>>(Wv, WvT);
        transpose_cvt_kernel<<<tg, tb, 0, stream>>>(Wo, WoT);
        const int nx4 = (int)(nX / 4);
        cvt_f32_bf16_kernel<<<(nx4 + 255) / 256, 256, 0, stream>>>(hs,  X_bf,  nx4);
        cvt_f32_bf16_kernel<<<(nx4 + 255) / 256, 256, 0, stream>>>(kst, KS_bf, nx4);
        cvt_f32_bf16_kernel<<<(nx4 + 255) / 256, 256, 0, stream>>>(vst, VS_bf, nx4);
    }

    // Stage 1: projections
    {
        const int strips = (B_ * SQ_ / 16) * (D_ / 64);     // 4096
        dim3 blk(32, 4);
        const float scale = 0.125f;                         // HD^-0.5
        proj_qkv_kernel<<<strips / 4, blk, 0, stream>>>(X_bf,  WqT, bq, Q_bf,  scale, 0);
        proj_qkv_kernel<<<strips / 4, blk, 0, stream>>>(KS_bf, WkT, bk, K_bf,  1.0f,  0);
        proj_qkv_kernel<<<strips / 4, blk, 0, stream>>>(VS_bf, WvT, bv, Vt_bf, 1.0f,  1);
    }

    // Stages 2-4 fused: scores + softmax + AV, one pass over weights
    {
        dim3 grd(SQ_ / 16, B_ * H_);                        // (128, 32)
        attn_fused_kernel<<<grd, 256, SMEM_BYTES, stream>>>(
            Q_bf, K_bf, Vt_bf, mask, weights, A_bf);
    }

    // Stage 5: output projection
    {
        const int strips = (B_ * SQ_ / 16) * (D_ / 64);
        dim3 blk(32, 4);
        oproj_kernel<<<strips / 4, blk, 0, stream>>>(A_bf, WoT, bo, out);
    }
}
